// ODConv2d_1623497638458
// MI455X (gfx1250) — matrix operand order repaired and checked by reading, not compiled, not run on hardware
//
#include <hip/hip_runtime.h>
#include <hip/hip_bf16.h>

// ---------------------------------------------------------------------------
// ODConv2d for MI455X (gfx1250): bf16 WMMA implicit-GEMM grouped conv,
// activations staged into LDS via the Tensor Data Mover (async DMA).
// B=32, CIN=COUT=64, KH=3, KNUM=4, ACH=16, H=W=128.
// ---------------------------------------------------------------------------

#define B_    32
#define CIN_  64
#define COUT_ 64
#define KH_   3
#define KNUM_ 4
#define ACH_  16
#define H_    128
#define W_    128
#define KTOT  576            // 9 taps * 64 channels  (K of the implicit GEMM)
#define WPITCH 584           // padded bf16 pitch of one filter row in LDS (73*8)
#define XPITCH 72            // padded bf16 channel pitch of activation tile
#define XCOLS  130           // 128 cols + 2 halo

typedef __attribute__((ext_vector_type(16))) __bf16 bf16x16;
typedef __attribute__((ext_vector_type(8)))  float  v8f;
typedef __attribute__((ext_vector_type(4)))  unsigned int u32x4;
typedef __attribute__((ext_vector_type(8)))  unsigned int u32x8;

union FragU { uint4 u[2]; bf16x16 v; };   // 32B fragment, 16B-aligned loads

// ---------------------------------------------------------------------------
// Kernel 1: global average pool  x(B,C,H,W) -> pooled(B*C)
// ---------------------------------------------------------------------------
__global__ __launch_bounds__(256) void pool_kernel(const float* __restrict__ x,
                                                   float* __restrict__ pooled) {
  __shared__ float red[256];
  const int bc = blockIdx.x;
  const float* px = x + (size_t)bc * (H_ * W_);
  float s = 0.f;
  for (int i = threadIdx.x; i < H_ * W_; i += 256) s += px[i];
  red[threadIdx.x] = s;
  __syncthreads();
  for (int off = 128; off > 0; off >>= 1) {
    if (threadIdx.x < off) red[threadIdx.x] += red[threadIdx.x + off];
    __syncthreads();
  }
  if (threadIdx.x == 0) pooled[bc] = red[0] * (1.0f / (H_ * W_));
}

// ---------------------------------------------------------------------------
// Kernel 2: attention heads (tiny: one thread per sample, one wave total)
// ---------------------------------------------------------------------------
__global__ __launch_bounds__(32) void attn_kernel(
    const float* __restrict__ pooled,
    const float* __restrict__ w1, const float* __restrict__ bn_g, const float* __restrict__ bn_b,
    const float* __restrict__ w_ch, const float* __restrict__ b_ch,
    const float* __restrict__ w_fl, const float* __restrict__ b_fl,
    const float* __restrict__ w_sp, const float* __restrict__ b_sp,
    const float* __restrict__ w_ke, const float* __restrict__ b_ke,
    float* __restrict__ chv, float* __restrict__ flv,
    float* __restrict__ spv, float* __restrict__ kev) {
  const int b = threadIdx.x;
  if (b >= B_) return;
  const float* pr = pooled + b * CIN_;
  float hfc[ACH_];
  const float bnscale = rsqrtf(1.0f + 1e-5f);
  for (int a = 0; a < ACH_; ++a) {
    float s = 0.f;
    for (int c = 0; c < CIN_; ++c) s += pr[c] * w1[a * CIN_ + c];
    s = s * (bn_g[a] * bnscale) + bn_b[a];
    hfc[a] = fmaxf(s, 0.f);
  }
  for (int c = 0; c < CIN_; ++c) {
    float s = b_ch[c];
    for (int a = 0; a < ACH_; ++a) s += hfc[a] * w_ch[c * ACH_ + a];
    float g = 1.f / (1.f + expf(-0.5f * s));
    chv[b * CIN_ + c] = fminf(fmaxf(g, 0.01f), 0.99f);
  }
  for (int o = 0; o < COUT_; ++o) {
    float s = b_fl[o];
    for (int a = 0; a < ACH_; ++a) s += hfc[a] * w_fl[o * ACH_ + a];
    float g = 1.f / (1.f + expf(-0.5f * s));
    flv[b * COUT_ + o] = fminf(fmaxf(g, 0.01f), 0.99f);
  }
  for (int p = 0; p < KH_ * KH_; ++p) {
    float s = b_sp[p];
    for (int a = 0; a < ACH_; ++a) s += hfc[a] * w_sp[p * ACH_ + a];
    float g = 1.f / (1.f + expf(-0.5f * s));
    spv[b * 9 + p] = fminf(fmaxf(g, 0.01f), 0.99f);
  }
  float l[KNUM_];
  float mx = -3.0e38f;
  for (int n = 0; n < KNUM_; ++n) {
    float s = b_ke[n];
    for (int a = 0; a < ACH_; ++a) s += hfc[a] * w_ke[n * ACH_ + a];
    l[n] = 0.5f * s;
    mx = fmaxf(mx, l[n]);
  }
  float den = 0.f;
  for (int n = 0; n < KNUM_; ++n) { l[n] = expf(l[n] - mx); den += l[n]; }
  for (int n = 0; n < KNUM_; ++n) kev[b * KNUM_ + n] = l[n] / den;
}

// ---------------------------------------------------------------------------
// Kernel 3: per-sample aggregated filters -> bf16, layout [b][o][p=kh*3+kw][i]
// ---------------------------------------------------------------------------
__global__ __launch_bounds__(256) void agg_kernel(const float* __restrict__ weight,
                                                  const float* __restrict__ spv,
                                                  const float* __restrict__ kev,
                                                  __bf16* __restrict__ agg) {
  const int e = blockIdx.x * 256 + threadIdx.x;   // exact: B*COUT*9*CIN
  const int i = e & 63;
  const int p = (e >> 6) % 9;
  const int o = (e / (64 * 9)) & 63;
  const int b = e / (64 * 9 * 64);
  float s = 0.f;
  for (int n = 0; n < KNUM_; ++n)
    s += kev[b * KNUM_ + n] * weight[(((size_t)n * COUT_ + o) * CIN_ + i) * 9 + p];
  agg[e] = (__bf16)(spv[b * 9 + p] * s * (1.0f / KNUM_));
}

// ---------------------------------------------------------------------------
// Kernel 4: y[b][h][w][c] = bf16(x[b][c][h][w] * ch[b][c])   (NCHW -> NHWC)
// ---------------------------------------------------------------------------
__global__ __launch_bounds__(256) void nhwc_kernel(const float* __restrict__ x,
                                                   const float* __restrict__ chv,
                                                   __bf16* __restrict__ y) {
  __shared__ float tile[CIN_][33];   // 33 pitch: conflict-free transpose
  const int blk = blockIdx.x;
  const int w0 = (blk & 3) * 32;
  const int h  = (blk >> 2) & 127;
  const int b  = blk >> 9;
  for (int idx = threadIdx.x; idx < CIN_ * 32; idx += 256) {
    const int c = idx >> 5, w = idx & 31;
    tile[c][w] = x[(((size_t)b * CIN_ + c) * H_ + h) * W_ + w0 + w] * chv[b * CIN_ + c];
  }
  __syncthreads();
  for (int idx = threadIdx.x; idx < 32 * CIN_; idx += 256) {
    const int w = idx >> 6, c = idx & 63;
    y[(((size_t)b * H_ + h) * W_ + w0 + w) * CIN_ + c] = (__bf16)tile[c][w];
  }
}

// ---------------------------------------------------------------------------
// Kernel 5: implicit-GEMM conv row.  Block = (b, h).  8 waves of wave32.
//   GEMM: M=COUT(64) x N=W(128) x K=576 on v_wmma_f32_16x16x32_bf16.
//   Activations DMA'd into LDS by the Tensor Data Mover with pad-insertion
//   (pad_interval=32 DWORDs, pad_amount=4 DWORDs) reproducing XPITCH=72.
// ---------------------------------------------------------------------------
__global__ __launch_bounds__(256) void conv_kernel(const __bf16* __restrict__ y,
                                                   const __bf16* __restrict__ agg,
                                                   const float* __restrict__ flv,
                                                   float* __restrict__ out) {
  extern __shared__ char smem[];
  __bf16* wlds = (__bf16*)smem;                               // [64][WPITCH]
  __bf16* xlds = (__bf16*)(smem + COUT_ * WPITCH * 2);        // [3][130][XPITCH]
  const int b = blockIdx.x >> 7;
  const int h = blockIdx.x & 127;
  const int tid = threadIdx.x;

  // ---- Stage activations via TDM (wave 0 only; scalar branch so waves 1..7
  //      truly skip the EXEC-ignoring tensor op). D# per ISA 08_async_tensor.
  const int waveid_s = __builtin_amdgcn_readfirstlane(tid >> 5);
  if (waveid_s == 0) {
    // group1: workgroup_mask=0, data_size=2B(code1), pad_enable,
    // pad_interval=code4 (32 DWORDs), pad_amount=code3 (4 DWORDs),
    // tensor_dim0=8192, tensor_dim1=1, tile_dim0=8192, tile_dim1=1,
    // tensor_dim0_stride=8192.
    const u32x8 g1 = {0x07110000u, 0x20000000u, 0x00010000u, 0x20000000u,
                      0x00000001u, 0x00002000u, 0u, 0u};
    for (int r = 0; r < 3; ++r) {
      const int hr = h + r - 1;
      if ((unsigned)hr < (unsigned)H_) {
        const unsigned long long ga = (unsigned long long)(size_t)
            (y + (((size_t)b * H_ + hr) * W_) * CIN_);
        const unsigned ldsoff =
            (unsigned)(size_t)(xlds + ((size_t)r * XCOLS + 1) * XPITCH);
        const u32x4 g0 = {1u,                       // count=1 descriptor
                          ldsoff,                   // lds_addr [63:32]
                          (unsigned)ga,             // global_addr low
                          ((unsigned)(ga >> 32) & 0x01FFFFFFu) |
                              0x80000000u};         // addr[56:32] | type=2
        asm volatile("tensor_load_to_lds %0, %1" :: "s"(g0), "s"(g1) : "memory");
      }
    }
    __builtin_amdgcn_s_wait_tensorcnt(0);
  }

  // ---- Stage filters (this sample): 64 rows x 576 bf16, 16B vector copies.
  {
    const uint4* src = (const uint4*)(agg + (size_t)b * COUT_ * KTOT);
    for (int idx = tid; idx < COUT_ * (KTOT / 8); idx += 256) {  // 4608
      const int o = idx / (KTOT / 8);
      const int s = idx - o * (KTOT / 8);
      *(uint4*)(wlds + o * WPITCH + s * 8) = src[idx];
    }
  }
  // ---- Zero-fill out-of-bounds halo rows and halo columns 0/129.
  {
    for (int r = 0; r < 3; ++r) {
      const int hr = h + r - 1;
      if ((unsigned)hr >= (unsigned)H_) {
        for (int idx = tid; idx < W_ * 8; idx += 256) {
          const int w = idx >> 3, s = idx & 7;
          *(uint4*)(xlds + ((size_t)r * XCOLS + (w + 1)) * XPITCH + s * 8) =
              make_uint4(0u, 0u, 0u, 0u);
        }
      }
    }
    for (int idx = tid; idx < 48; idx += 256) {
      const int r = idx / 16, e = idx & 15;
      const int col = (e < 8) ? 0 : (XCOLS - 1);
      const int s = e & 7;
      *(uint4*)(xlds + ((size_t)r * XCOLS + col) * XPITCH + s * 8) =
          make_uint4(0u, 0u, 0u, 0u);
    }
  }
  __syncthreads();

  const int wave = tid >> 5, lane = tid & 31;
  const int mbase = (wave >> 1) << 4;    // 0,16,32,48 : output-channel tile
  const int wbase = (wave & 1) << 6;     // 0,64       : spatial half-row
  const int lm = lane & 15;
  const int g  = lane >> 4;              // lane half (ISA fragment layouts)

  v8f acc0 = {}, acc1 = {}, acc2 = {}, acc3 = {};

#pragma unroll
  for (int p = 0; p < 9; ++p) {          // filter tap (kh,kw)
    const int kh = p / 3, kw = p - kh * 3;
    const __bf16* xr = xlds + (size_t)kh * XCOLS * XPITCH;
#pragma unroll
    for (int half = 0; half < 2; ++half) {   // channel halves 0..31 / 32..63
      // A fragment (16x32 bf16): lane half g holds K = 16g..16g+15
      const __bf16* arow = wlds + (mbase + lm) * WPITCH + p * CIN_ + half * 32;
      FragU a;
      a.u[0] = *(const uint4*)(arow + 16 * g);
      a.u[1] = *(const uint4*)(arow + 16 * g + 8);
      // B fragments (32x16 bf16): lane holds 16 contiguous K at offset 16g
      const __bf16* xbase = xr + half * 32 + g * 16;
      const __bf16* p0 = xbase + (size_t)(wbase +  0 + lm + kw) * XPITCH;
      const __bf16* p1 = xbase + (size_t)(wbase + 16 + lm + kw) * XPITCH;
      const __bf16* p2 = xbase + (size_t)(wbase + 32 + lm + kw) * XPITCH;
      const __bf16* p3 = xbase + (size_t)(wbase + 48 + lm + kw) * XPITCH;
      FragU f0, f1, f2, f3;
      f0.u[0] = ((const uint4*)p0)[0]; f0.u[1] = ((const uint4*)p0)[1];
      f1.u[0] = ((const uint4*)p1)[0]; f1.u[1] = ((const uint4*)p1)[1];
      f2.u[0] = ((const uint4*)p2)[0]; f2.u[1] = ((const uint4*)p2)[1];
      f3.u[0] = ((const uint4*)p3)[0]; f3.u[1] = ((const uint4*)p3)[1];
      acc0 = __builtin_amdgcn_wmma_f32_16x16x32_bf16(false, a.v, false, f0.v,
                                                     (short)0, acc0, false, false);
      acc1 = __builtin_amdgcn_wmma_f32_16x16x32_bf16(false, a.v, false, f1.v,
                                                     (short)0, acc1, false, false);
      acc2 = __builtin_amdgcn_wmma_f32_16x16x32_bf16(false, a.v, false, f2.v,
                                                     (short)0, acc2, false, false);
      acc3 = __builtin_amdgcn_wmma_f32_16x16x32_bf16(false, a.v, false, f3.v,
                                                     (short)0, acc3, false, false);
    }
  }

  // Epilogue: D layout (ISA 7.12.2): VGPR r, lane -> M = r + 8g, N = lane&15.
  const int og = mbase + (g << 3);
#pragma unroll
  for (int r = 0; r < 8; ++r) {
    const int o = og + r;
    const float f = flv[b * COUT_ + o];
    float* orow = out + (((size_t)b * COUT_ + o) * H_ + h) * W_;
    orow[wbase +  0 + lm] = acc0[r] * f;
    orow[wbase + 16 + lm] = acc1[r] * f;
    orow[wbase + 32 + lm] = acc2[r] * f;
    orow[wbase + 48 + lm] = acc3[r] * f;
  }
}

// ---------------------------------------------------------------------------
extern "C" void kernel_launch(void* const* d_in, const int* in_sizes, int n_in,
                              void* d_out, int out_size, void* d_ws, size_t ws_size,
                              hipStream_t stream) {
  (void)in_sizes; (void)n_in; (void)out_size; (void)ws_size;
  const float* x     = (const float*)d_in[0];
  const float* w1    = (const float*)d_in[1];
  const float* bn_g  = (const float*)d_in[2];
  const float* bn_b  = (const float*)d_in[3];
  const float* w_ch  = (const float*)d_in[4];
  const float* b_ch  = (const float*)d_in[5];
  const float* w_fl  = (const float*)d_in[6];
  const float* b_fl  = (const float*)d_in[7];
  const float* w_sp  = (const float*)d_in[8];
  const float* b_sp  = (const float*)d_in[9];
  const float* w_ke  = (const float*)d_in[10];
  const float* b_ke  = (const float*)d_in[11];
  const float* weight = (const float*)d_in[12];
  float* out = (float*)d_out;

  // Workspace layout
  float* pooled = (float*)d_ws;                        // 2048 f32
  float* chv = pooled + B_ * CIN_;                     // 2048 f32
  float* flv = chv + B_ * CIN_;                        // 2048 f32
  float* spv = flv + B_ * COUT_;                       // 288  f32
  float* kev = spv + B_ * 9;                           // 128  f32
  __bf16* agg = (__bf16*)((char*)d_ws + 26624);        // 1,179,648 bf16 (2.36MB)
  __bf16* y = (__bf16*)((char*)d_ws + 26624 + (size_t)B_ * COUT_ * 9 * CIN_ * 2); // 64MB

  const int CONV_SMEM = COUT_ * WPITCH * 2 + 3 * XCOLS * XPITCH * 2;  // 130912 B
  (void)hipFuncSetAttribute((const void*)conv_kernel,
                            hipFuncAttributeMaxDynamicSharedMemorySize, CONV_SMEM);

  pool_kernel<<<B_ * CIN_, 256, 0, stream>>>(x, pooled);
  attn_kernel<<<1, 32, 0, stream>>>(pooled, w1, bn_g, bn_b, w_ch, b_ch, w_fl, b_fl,
                                    w_sp, b_sp, w_ke, b_ke, chv, flv, spv, kev);
  agg_kernel<<<(B_ * COUT_ * 9 * CIN_) / 256, 256, 0, stream>>>(weight, spv, kev, agg);
  nhwc_kernel<<<B_ * H_ * (W_ / 32), 256, 0, stream>>>(x, chv, y);
  conv_kernel<<<B_ * H_, 256, CONV_SMEM, stream>>>(y, agg, flv, out);
}
